// CausalSelfAttention_56530359550854
// MI455X (gfx1250) — matrix operand order repaired and checked by reading, not compiled, not run on hardware
//
#include <hip/hip_runtime.h>
#include <hip/hip_bf16.h>
#include <math.h>

// ---------------------------------------------------------------------------
// CDNA5 (gfx1250) causal self-attention: wave32 + v_wmma_f32_16x16x32_f16
// B=2, T=2048, C=1024, NH=16, HD=64
// ---------------------------------------------------------------------------

typedef __attribute__((ext_vector_type(16))) _Float16 v16h;
typedef __attribute__((ext_vector_type(8)))  _Float16 v8h;
typedef __attribute__((ext_vector_type(8)))  float    v8f;

#define CB  2
#define CT  2048
#define CC  1024
#define CNH 16
#define CHD 64

#define WMMA(a, b, c) \
  __builtin_amdgcn_wmma_f32_16x16x32_f16(false, (a), false, (b), (short)0, (c), false, false)

// A-fragment (16x32 f16): lane holds row (row0 + lane%16);
// lanes 0-15: K = {0..7, 16..23}; lanes 16-31: K = {8..15, 24..31}
__device__ __forceinline__ v16h load_fragA(const _Float16* base, int row0, int colOff,
                                           int stride, int lane) {
  const _Float16* p = base + (size_t)(row0 + (lane & 15)) * stride + colOff + ((lane & 16) >> 1);
  v8h lo = *(const v8h*)(p);
  v8h hi = *(const v8h*)(p + 16);
  return __builtin_shufflevector(lo, hi, 0,1,2,3,4,5,6,7,8,9,10,11,12,13,14,15);
}

// B-fragment (32x16 f16): lane holds column (row0 + lane%16) of tile rows;
// lanes 0-15: K = {0..7, 16..23}; lanes 16-31: K = {8..15, 24..31}
__device__ __forceinline__ v16h load_fragB(const _Float16* base, int row0, int colOff,
                                           int stride, int lane) {
  const _Float16* p = base + (size_t)(row0 + (lane & 15)) * stride + colOff + ((lane & 16) >> 1);
  v8h lo = *(const v8h*)(p);
  v8h hi = *(const v8h*)(p + 16);
  return __builtin_shufflevector(lo, hi, 0,1,2,3,4,5,6,7,8,9,10,11,12,13,14,15);
}

__device__ __forceinline__ v8h pack8(float4 a, float4 b, float s) {
  v8h h;
  h[0]=(_Float16)(a.x*s); h[1]=(_Float16)(a.y*s); h[2]=(_Float16)(a.z*s); h[3]=(_Float16)(a.w*s);
  h[4]=(_Float16)(b.x*s); h[5]=(_Float16)(b.y*s); h[6]=(_Float16)(b.z*s); h[7]=(_Float16)(b.w*s);
  return h;
}

// ---------------------------------------------------------------------------
// GEMM: Cmat[M,N] = A[M,K] @ Bm[N,K]^T + bias[N]
// EPI==1: fused RoPE on q/k columns (cols < 2C), used for the QKV projection.
// Block: 256 threads (8 waves), tile 128x128, K-step 32.
// ---------------------------------------------------------------------------
#define LDA 40   // LDS row stride in halves (80B = 20 banks, conflict-free)

template<int EPI>
__global__ __launch_bounds__(256)
void gemm_wmma_kernel(const float* __restrict__ A, const float* __restrict__ Bm,
                      const float* __restrict__ bias, float* __restrict__ Cmat,
                      int M, int N, int K) {
  const int lane  = threadIdx.x & 31;
  const int wid   = threadIdx.x >> 5;
  const int waveM = wid & 3;          // 4 waves along M (32 rows each)
  const int waveN = wid >> 2;         // 2 waves along N (64 cols each)
  const int mBase = blockIdx.x * 128;
  const int nBase = blockIdx.y * 128;

  __shared__ _Float16 As[128][LDA];
  __shared__ _Float16 Bs[128][LDA];

  v8f acc[2][4];
#pragma unroll
  for (int mi = 0; mi < 2; mi++)
#pragma unroll
    for (int ni = 0; ni < 4; ni++)
#pragma unroll
      for (int r = 0; r < 8; r++) acc[mi][ni][r] = 0.0f;

  const int t     = threadIdx.x;
  const int ldRow = t >> 1;           // 0..127
  const int ldCol = (t & 1) * 16;     // 0 or 16

  for (int k0 = 0; k0 < K; k0 += 32) {
    const float* ga = A  + (size_t)(mBase + ldRow) * K + k0 + ldCol;
    const float* gb = Bm + (size_t)(nBase + ldRow) * K + k0 + ldCol;
    float4 a0 = *(const float4*)(ga);     float4 a1 = *(const float4*)(ga + 4);
    float4 a2 = *(const float4*)(ga + 8); float4 a3 = *(const float4*)(ga + 12);
    float4 b0 = *(const float4*)(gb);     float4 b1 = *(const float4*)(gb + 4);
    float4 b2 = *(const float4*)(gb + 8); float4 b3 = *(const float4*)(gb + 12);
    *(v8h*)&As[ldRow][ldCol]     = pack8(a0, a1, 1.0f);
    *(v8h*)&As[ldRow][ldCol + 8] = pack8(a2, a3, 1.0f);
    *(v8h*)&Bs[ldRow][ldCol]     = pack8(b0, b1, 1.0f);
    *(v8h*)&Bs[ldRow][ldCol + 8] = pack8(b2, b3, 1.0f);
    __syncthreads();

    v16h afr[2];
#pragma unroll
    for (int mi = 0; mi < 2; mi++)
      afr[mi] = load_fragA(&As[0][0], waveM * 32 + mi * 16, 0, LDA, lane);
#pragma unroll
    for (int ni = 0; ni < 4; ni++) {
      v16h bfr = load_fragB(&Bs[0][0], waveN * 64 + ni * 16, 0, LDA, lane);
      acc[0][ni] = WMMA(afr[0], bfr, acc[0][ni]);
      acc[1][ni] = WMMA(afr[1], bfr, acc[1][ni]);
    }
    __syncthreads();
  }

  // Epilogue: bias (+ optional RoPE), fp32 store.
  const int rowHalf = (lane & 16) ? 8 : 0;
#pragma unroll
  for (int mi = 0; mi < 2; mi++) {
#pragma unroll
    for (int ni = 0; ni < 4; ni++) {
      const int row0 = mBase + waveM * 32 + mi * 16 + rowHalf;
      const int col  = nBase + waveN * 64 + ni * 16 + (lane & 15);
      const float bv = bias[col];
      float rope = 1.0f;
      if (EPI == 1) {
        // q: col in [0,C), k: col in [C,2C)  -> apply cos(ang)+sin(ang)
        const int d = col & (CHD - 1);
        const int j = d & (CHD / 2 - 1);
        const float theta = __powf(10000.0f, -(float)j * (2.0f / (float)CHD));
        if (col < 2 * CC) rope = 0.0f; // marker; recomputed per-row below
      }
#pragma unroll
      for (int r = 0; r < 8; r++) {
        const int row = row0 + r;
        float v = acc[mi][ni][r] + bv;
        if (EPI == 1 && col < 2 * CC) {
          const int d = col & (CHD - 1);
          const int j = d & (CHD / 2 - 1);
          const float theta = __powf(10000.0f, -(float)j * (2.0f / (float)CHD));
          const float ang = (float)((row & (CT - 1)) + 1) * theta;
          v *= (__cosf(ang) + __sinf(ang));
        }
        Cmat[(size_t)row * N + col] = v;
      }
      (void)rope;
    }
  }
}

// ---------------------------------------------------------------------------
// Flash attention: grid (T/64 q-tiles, B*NH), 128 threads = 4 waves.
// qkv layout [B*T, 3C]; q/k already RoPE'd; 1/sqrt(HD) folded into Q staging.
// Output y1 [B*T, C] with col = h*HD + d.
// ---------------------------------------------------------------------------
#define LDT 72   // LDS row stride in halves (144B = 36 banks, conflict-free)

__global__ __launch_bounds__(128)
void attn_wmma_kernel(const float* __restrict__ qkv, float* __restrict__ y1) {
  const int lane  = threadIdx.x & 31;
  const int wid   = threadIdx.x >> 5;        // 0..3, wave owns 16 q rows
  const int bh    = blockIdx.y;
  const int b     = bh >> 4;
  const int h     = bh & 15;
  const int qTile = blockIdx.x;
  const int qBase = qTile * 64;
  const int rowHalf = (lane & 16) ? 8 : 0;

  __shared__ _Float16 Qs[64][LDT];           // [q row][d]
  __shared__ _Float16 Ks[64][LDT];           // [key][d]
  __shared__ _Float16 Vt[64][LDT];           // transposed: [d][key]
  __shared__ _Float16 Ps[4][16][LDT];        // per-wave P staging [row][key]

  const int t    = threadIdx.x;
  const int lrow = t >> 1;                   // 0..63
  const int lc0  = (t & 1) * 32;             // 0 or 32

  // Stage Q tile (scaled by 1/sqrt(HD) = 1/8)
  {
    const float* g = qkv + (size_t)(b * CT + qBase + lrow) * (3 * CC) + h * CHD + lc0;
#pragma unroll
    for (int i = 0; i < 4; i++) {
      float4 f0 = *(const float4*)(g + i * 8);
      float4 f1 = *(const float4*)(g + i * 8 + 4);
      *(v8h*)&Qs[lrow][lc0 + i * 8] = pack8(f0, f1, 0.125f);
    }
  }
  __syncthreads();

  v16h qfr[2];
  qfr[0] = load_fragA(&Qs[0][0], wid * 16, 0,  LDT, lane);
  qfr[1] = load_fragA(&Qs[0][0], wid * 16, 32, LDT, lane);

  v8f o[4];
#pragma unroll
  for (int ni = 0; ni < 4; ni++)
#pragma unroll
    for (int r = 0; r < 8; r++) o[ni][r] = 0.0f;
  float m[8], l[8];
#pragma unroll
  for (int r = 0; r < 8; r++) { m[r] = -3.0e38f; l[r] = 0.0f; }

  for (int j = 0; j <= qTile; j++) {
    __syncthreads();
    // Stage K tile (row-major) and V tile (transposed)
    {
      const size_t rb = (size_t)(b * CT + j * 64 + lrow) * (3 * CC) + h * CHD + lc0;
      const float* gk = qkv + rb + CC;
      const float* gv = qkv + rb + 2 * CC;
#pragma unroll
      for (int i = 0; i < 4; i++) {
        float4 f0 = *(const float4*)(gk + i * 8);
        float4 f1 = *(const float4*)(gk + i * 8 + 4);
        *(v8h*)&Ks[lrow][lc0 + i * 8] = pack8(f0, f1, 1.0f);
      }
#pragma unroll
      for (int i = 0; i < 8; i++) {
        float4 f = *(const float4*)(gv + i * 4);
        Vt[lc0 + i * 4 + 0][lrow] = (_Float16)f.x;
        Vt[lc0 + i * 4 + 1][lrow] = (_Float16)f.y;
        Vt[lc0 + i * 4 + 2][lrow] = (_Float16)f.z;
        Vt[lc0 + i * 4 + 3][lrow] = (_Float16)f.w;
      }
    }
    __syncthreads();

    // S = (Q * 1/8) @ K^T   (16 rows x 64 keys per wave)
    v8f s[4];
#pragma unroll
    for (int ni = 0; ni < 4; ni++) {
#pragma unroll
      for (int r = 0; r < 8; r++) s[ni][r] = 0.0f;
      v16h k0 = load_fragB(&Ks[0][0], ni * 16, 0,  LDT, lane);
      v16h k1 = load_fragB(&Ks[0][0], ni * 16, 32, LDT, lane);
      s[ni] = WMMA(qfr[0], k0, s[ni]);
      s[ni] = WMMA(qfr[1], k1, s[ni]);
    }

    // Causal mask (only the diagonal key tile needs it)
    if (j == qTile) {
#pragma unroll
      for (int ni = 0; ni < 4; ni++)
#pragma unroll
        for (int r = 0; r < 8; r++) {
          const int qr = wid * 16 + r + rowHalf;       // row within tile
          const int kc = ni * 16 + (lane & 15);        // key within tile
          if (kc > qr) s[ni][r] = -3.0e38f;
        }
    }

    // Online softmax (rows live in 16-lane groups; reduce with xor shuffles)
#pragma unroll
    for (int r = 0; r < 8; r++) {
      float mx = s[0][r];
      mx = fmaxf(mx, s[1][r]); mx = fmaxf(mx, s[2][r]); mx = fmaxf(mx, s[3][r]);
#pragma unroll
      for (int off = 1; off < 16; off <<= 1) mx = fmaxf(mx, __shfl_xor(mx, off, 32));
      const float mnew = fmaxf(m[r], mx);
      float ps = 0.0f;
#pragma unroll
      for (int ni = 0; ni < 4; ni++) {
        const float p = __expf(s[ni][r] - mnew);
        s[ni][r] = p;
        ps += p;
      }
#pragma unroll
      for (int off = 1; off < 16; off <<= 1) ps += __shfl_xor(ps, off, 32);
      const float alpha = __expf(m[r] - mnew);
      l[r] = l[r] * alpha + ps;
      m[r] = mnew;
#pragma unroll
      for (int ni = 0; ni < 4; ni++) o[ni][r] *= alpha;
    }

    // Stage P (C-layout -> LDS row-major) so it can be re-read in A-layout
#pragma unroll
    for (int ni = 0; ni < 4; ni++)
#pragma unroll
      for (int r = 0; r < 8; r++)
        Ps[wid][r + rowHalf][ni * 16 + (lane & 15)] = (_Float16)s[ni][r];
    __syncthreads();

    // O += P @ V
    v16h pf0 = load_fragA(&Ps[wid][0][0], 0, 0,  LDT, lane);
    v16h pf1 = load_fragA(&Ps[wid][0][0], 0, 32, LDT, lane);
#pragma unroll
    for (int ni = 0; ni < 4; ni++) {
      v16h v0 = load_fragB(&Vt[0][0], ni * 16, 0,  LDT, lane);
      v16h v1 = load_fragB(&Vt[0][0], ni * 16, 32, LDT, lane);
      o[ni] = WMMA(pf0, v0, o[ni]);
      o[ni] = WMMA(pf1, v1, o[ni]);
    }
  }

  // Normalize and store: y1[b*T + row][h*64 + d]
#pragma unroll
  for (int r = 0; r < 8; r++) {
    const float invl = 1.0f / l[r];
    const int row = qBase + wid * 16 + r + rowHalf;
    float* dst = y1 + (size_t)(b * CT + row) * CC + h * CHD;
#pragma unroll
    for (int ni = 0; ni < 4; ni++)
      dst[ni * 16 + (lane & 15)] = o[ni][r] * invl;
  }
}

// ---------------------------------------------------------------------------
// Launch: qkv GEMM (+rope) -> flash attention -> output projection
// ---------------------------------------------------------------------------
extern "C" void kernel_launch(void* const* d_in, const int* in_sizes, int n_in,
                              void* d_out, int out_size, void* d_ws, size_t ws_size,
                              hipStream_t stream) {
  const float* x      = (const float*)d_in[0];
  const float* W_attn = (const float*)d_in[1];
  const float* b_attn = (const float*)d_in[2];
  const float* W_proj = (const float*)d_in[3];
  const float* b_proj = (const float*)d_in[4];
  float* out = (float*)d_out;

  float* qkv = (float*)d_ws;                              // [4096, 3072] = 48 MB
  float* y1  = qkv + (size_t)(CB * CT) * (3 * CC);        // [4096, 1024] = 16 MB

  const int M = CB * CT;                                  // 4096

  // 1) qkv = x @ W_attn^T + b_attn, with fused RoPE on q/k
  gemm_wmma_kernel<1><<<dim3(M / 128, (3 * CC) / 128), dim3(256), 0, stream>>>(
      x, W_attn, b_attn, qkv, M, 3 * CC, CC);

  // 2) flash attention per (q-tile, b*h)
  attn_wmma_kernel<<<dim3(CT / 64, CB * CNH), dim3(128), 0, stream>>>(qkv, y1);

  // 3) out = y1 @ W_proj^T + b_proj
  gemm_wmma_kernel<0><<<dim3(M / 128, CC / 128), dim3(256), 0, stream>>>(
      y1, W_proj, b_proj, out, M, CC, CC);

  (void)in_sizes; (void)n_in; (void)out_size; (void)ws_size;
}
